// SelfAttentionLayerWavelet_71055938945730
// MI455X (gfx1250) — matrix operand order repaired and checked by reading, not compiled, not run on hardware
//
#include <hip/hip_runtime.h>

typedef __attribute__((ext_vector_type(16))) _Float16 v16h;
typedef __attribute__((ext_vector_type(8)))  float    v8f;

// ---- problem constants ----
constexpr int kB   = 8;
constexpr int kT   = 24;
constexpr int kN   = 2048;
constexpr int kC   = 64;
constexpr int kH   = 8;
constexpr int kS   = 4;
constexpr int kBT  = kB * kT;            // 192
constexpr int kBTN = kBT * kN;           // 393216
constexpr long long kBTNC = (long long)kBTN * kC; // 25165824
constexpr int kKBIG = kS * kN;           // 8192

// =====================================================================
// WMMA fragment loaders (CDNA5 16x16x32 f16 layouts, wave32)
// =====================================================================
__device__ __forceinline__ v16h load_frag_a(const _Float16* __restrict__ A,
                                            int lda, int r0, int k0, int lane) {
  // A is [M][K] row-major f16. lane&15 = row; lanes>=16 take K 8..15 / 24..31.
  int row = r0 + (lane & 15);
  int kb  = k0 + ((lane >> 4) << 3);
  const _Float16* p = A + (size_t)row * lda + kb;
  union { v16h v; float4 q[2]; } u;
  u.q[0] = *(const float4*)(p);        // K = kb .. kb+7
  u.q[1] = *(const float4*)(p + 16);   // K = kb+16 .. kb+23
  return u.v;
}

__device__ __forceinline__ v16h load_frag_b(const _Float16* __restrict__ Bt,
                                            int ldb, int c0, int k0, int lane) {
  // Bt is [N][K] row-major f16 (B transposed). lane&15 = col; K block by lane>=16.
  int col = c0 + (lane & 15);
  int kb  = k0 + ((lane >> 4) << 3);
  const _Float16* p = Bt + (size_t)col * ldb + kb;
  union { v16h v; float4 q[2]; } u;
  u.q[0] = *(const float4*)(p);        // K = kb .. kb+7
  u.q[1] = *(const float4*)(p + 16);   // K = kb+16 .. kb+23
  return u.v;
}

// =====================================================================
// Generic LDS-free WMMA GEMM: D = A * Bt^T (+bias)(relu), f16 in, f32 acc
// Operands are L2-resident by construction (192MB L2 on MI455X).
// Block = GWM*GWN waves; each wave computes (WM*16) x (WN*16).
// __launch_bounds__(256,1): full VGPR budget -> spill-free inner loop.
// =====================================================================
template<int WM, int WN, int GWM, int GWN, bool BIAS, bool RELU, bool O32, bool O16>
__global__ __launch_bounds__(256, 1)
void gemm_wmma(const _Float16* __restrict__ A, int lda, long long sA,
               const _Float16* __restrict__ Bt, int ldb, long long sB,
               float* __restrict__ D32, _Float16* __restrict__ D16,
               int ldd, long long sD,
               const float* __restrict__ bias, int K) {
  const int lane = threadIdx.x & 31;
  const int wave = threadIdx.x >> 5;
  const int wm = wave / GWN;
  const int wn = wave % GWN;
  const int row0 = (blockIdx.x * GWM + wm) * (WM * 16);
  const int col0 = (blockIdx.y * GWN + wn) * (WN * 16);

  A  += (size_t)blockIdx.z * sA;
  Bt += (size_t)blockIdx.z * sB;

  const v8f zero = {0.f,0.f,0.f,0.f,0.f,0.f,0.f,0.f};
  v8f acc[WM][WN];
#pragma unroll
  for (int i = 0; i < WM; ++i)
#pragma unroll
    for (int j = 0; j < WN; ++j) acc[i][j] = zero;

  for (int k0 = 0; k0 < K; k0 += 32) {
    v16h af[WM], bf[WN];
#pragma unroll
    for (int i = 0; i < WM; ++i) af[i] = load_frag_a(A, lda, row0 + i * 16, k0, lane);
#pragma unroll
    for (int j = 0; j < WN; ++j) bf[j] = load_frag_b(Bt, ldb, col0 + j * 16, k0, lane);
#pragma unroll
    for (int i = 0; i < WM; ++i)
#pragma unroll
      for (int j = 0; j < WN; ++j)
        acc[i][j] = __builtin_amdgcn_wmma_f32_16x16x32_f16(
            false, af[i], false, bf[j], (short)0, acc[i][j], false, false);
  }

  // epilogue
#pragma unroll
  for (int i = 0; i < WM; ++i) {
#pragma unroll
    for (int j = 0; j < WN; ++j) {
      const int col   = col0 + j * 16 + (lane & 15);
      const int rbase = row0 + i * 16 + ((lane >> 4) << 3);
      float bv;
      if constexpr (BIAS) bv = bias[col];
#pragma unroll
      for (int e = 0; e < 8; ++e) {
        float v = acc[i][j][e];
        if constexpr (BIAS) v += bv;
        if constexpr (RELU) v = fmaxf(v, 0.f);
        const size_t idx = (size_t)blockIdx.z * sD + (size_t)(rbase + e) * ldd + col;
        if constexpr (O32) D32[idx] = v;
        if constexpr (O16) D16[idx] = (_Float16)v;
      }
    }
  }
}

// =====================================================================
// elementwise / layout kernels
// =====================================================================
__global__ void cvt_f32_f16(const float* __restrict__ in, _Float16* __restrict__ out, long long n) {
  long long i = (long long)blockIdx.x * blockDim.x + threadIdx.x;
  if (i < n) out[i] = (_Float16)in[i];
}

// Ah[i][s*2048+j] = phi[s][i][j]
__global__ void build_ah(const float* __restrict__ phi, _Float16* __restrict__ Ah) {
  long long i = (long long)blockIdx.x * 256 + threadIdx.x;   // over S*N*N
  long long j = i & (kN - 1);
  long long r = (i >> 11) & (kN - 1);
  long long s = i >> 22;
  Ah[r * kKBIG + s * kN + j] = (_Float16)phi[i];
}

// Bth[n][s*2048+j] = diag_w[s][j] * phi_inv[s][j][n]   (tiled transpose)
__global__ void build_bth(const float* __restrict__ phi_inv, const float* __restrict__ diag_w,
                          _Float16* __restrict__ Bth) {
  __shared__ float tile[32][33];
  const int s  = blockIdx.z;
  const int n0 = blockIdx.x * 32;
  const int j0 = blockIdx.y * 32;
#pragma unroll
  for (int kk = 0; kk < 4; ++kk) {
    int j = j0 + threadIdx.y + kk * 8;
    tile[threadIdx.y + kk * 8][threadIdx.x] =
        phi_inv[((size_t)s * kN + j) * kN + n0 + threadIdx.x] * diag_w[s * kN + j];
  }
  __syncthreads();
#pragma unroll
  for (int kk = 0; kk < 4; ++kk) {
    int n = n0 + threadIdx.y + kk * 8;
    Bth[(size_t)n * kKBIG + (size_t)s * kN + j0 + threadIdx.x] =
        (_Float16)tile[threadIdx.x][threadIdx.y + kk * 8];
  }
}

// xt[bt][c][n] = x[bt][n][c]   (tiled transpose, f32 -> f16)
__global__ void build_xt(const float* __restrict__ x, _Float16* __restrict__ xt) {
  __shared__ float tile[32][33];
  const int bt = blockIdx.z;
  const int n0 = blockIdx.x * 32;
  const int c0 = blockIdx.y * 32;
#pragma unroll
  for (int kk = 0; kk < 4; ++kk) {
    int n = n0 + threadIdx.y + kk * 8;
    tile[threadIdx.y + kk * 8][threadIdx.x] =
        x[((size_t)bt * kN + n) * kC + c0 + threadIdx.x];
  }
  __syncthreads();
#pragma unroll
  for (int kk = 0; kk < 4; ++kk) {
    int c = c0 + threadIdx.y + kk * 8;
    xt[((size_t)bt * kC + c) * kN + n0 + threadIdx.x] =
        (_Float16)tile[threadIdx.x][threadIdx.y + kk * 8];
  }
}

// =====================================================================
// linear attention (performer-style), qkv f16 [rows][192]
// spatial: one block per (bt, head), L = N = 2048
// =====================================================================
__global__ void attn_spatial(const _Float16* __restrict__ qkvh, _Float16* __restrict__ outcat) {
  const int bt   = blockIdx.x;
  const int h    = blockIdx.y;
  const int tid  = threadIdx.x;
  const int lane = tid & 31;
  const int wave = tid >> 5;

  float kvs[8][8];
  float ksum[8];
#pragma unroll
  for (int i = 0; i < 8; ++i) {
    ksum[i] = 0.f;
#pragma unroll
    for (int j = 0; j < 8; ++j) kvs[i][j] = 0.f;
  }

  for (int n = tid; n < kN; n += 256) {
    const _Float16* row = qkvh + ((size_t)bt * kN + n) * 192;
    float kk[8], vv[8];
#pragma unroll
    for (int d = 0; d < 8; ++d) {
      kk[d] = (float)row[64 + h * 8 + d];
      vv[d] = (float)row[128 + h * 8 + d];
    }
    float s = 0.f;
#pragma unroll
    for (int d = 0; d < 8; ++d) s += kk[d] * kk[d];
    float inv = 1.0f / fmaxf(sqrtf(s), 1e-12f);
#pragma unroll
    for (int d = 0; d < 8; ++d) kk[d] *= inv;
#pragma unroll
    for (int i = 0; i < 8; ++i) {
      ksum[i] += kk[i];
#pragma unroll
      for (int j = 0; j < 8; ++j) kvs[i][j] += kk[i] * vv[j];
    }
  }

  __shared__ float s_part[8][72];
  __shared__ float s_kvs[8][8];
  __shared__ float s_ksum[8];
#pragma unroll
  for (int m = 0; m < 72; ++m) {
    float v = (m < 64) ? kvs[m >> 3][m & 7] : ksum[m - 64];
#pragma unroll
    for (int o = 16; o > 0; o >>= 1) v += __shfl_down(v, o);
    if (lane == 0) s_part[wave][m] = v;
  }
  __syncthreads();
  if (tid < 72) {
    float s = 0.f;
#pragma unroll
    for (int w = 0; w < 8; ++w) s += s_part[w][tid];
    if (tid < 64) s_kvs[tid >> 3][tid & 7] = s;
    else          s_ksum[tid - 64] = s;
  }
  __syncthreads();

  const float L = (float)kN;
  for (int n = tid; n < kN; n += 256) {
    const _Float16* row = qkvh + ((size_t)bt * kN + n) * 192;
    float qq[8], vv[8];
#pragma unroll
    for (int d = 0; d < 8; ++d) {
      qq[d] = (float)row[h * 8 + d];
      vv[d] = (float)row[128 + h * 8 + d];
    }
    float s = 0.f;
#pragma unroll
    for (int d = 0; d < 8; ++d) s += qq[d] * qq[d];
    float inv = 1.0f / fmaxf(sqrtf(s), 1e-12f);
#pragma unroll
    for (int d = 0; d < 8; ++d) qq[d] *= inv;

    float den = L;
#pragma unroll
    for (int i = 0; i < 8; ++i) den += qq[i] * s_ksum[i];
    float invden = 1.0f / fmaxf(den, 1e-5f);

    _Float16* orow = outcat + ((size_t)bt * kN + n) * 128 + h * 8;
#pragma unroll
    for (int d = 0; d < 8; ++d) {
      float num = L * vv[d];
#pragma unroll
      for (int i = 0; i < 8; ++i) num += qq[i] * s_kvs[i][d];
      orow[d] = (_Float16)(num * invden);
    }
  }
}

// temporal: one block (192 threads = (t,h)) per (b, n), L = T = 24
__global__ void attn_temporal(const _Float16* __restrict__ qkvh, _Float16* __restrict__ outcat) {
  const int bn = blockIdx.x;
  const int b  = bn >> 11;          // / kN
  const int n  = bn & (kN - 1);
  const int tid = threadIdx.x;
  const int t = tid >> 3;
  const int h = tid & 7;

  __shared__ float s_k[24][8][8];
  __shared__ float s_v[24][8][8];
  __shared__ float s_kvs[8][8][8];
  __shared__ float s_ksum[8][8];

  if (tid < 192) {
    const _Float16* row = qkvh + (((size_t)(b * kT + t)) * kN + n) * 192;
    float kk[8];
    float s = 0.f;
#pragma unroll
    for (int d = 0; d < 8; ++d) {
      kk[d] = (float)row[64 + h * 8 + d];
      s_v[t][h][d] = (float)row[128 + h * 8 + d];
      s += kk[d] * kk[d];
    }
    float inv = 1.0f / fmaxf(sqrtf(s), 1e-12f);
#pragma unroll
    for (int d = 0; d < 8; ++d) s_k[t][h][d] = kk[d] * inv;
  }
  __syncthreads();
  if (tid < 64) {
    const int hh = tid >> 3, i = tid & 7;
    float sum = 0.f;
    float acc[8];
#pragma unroll
    for (int j = 0; j < 8; ++j) acc[j] = 0.f;
    for (int tt = 0; tt < kT; ++tt) {
      float kv = s_k[tt][hh][i];
      sum += kv;
#pragma unroll
      for (int j = 0; j < 8; ++j) acc[j] += kv * s_v[tt][hh][j];
    }
    s_ksum[hh][i] = sum;
#pragma unroll
    for (int j = 0; j < 8; ++j) s_kvs[hh][i][j] = acc[j];
  }
  __syncthreads();
  if (tid < 192) {
    const _Float16* row = qkvh + (((size_t)(b * kT + t)) * kN + n) * 192;
    float qq[8];
    float s = 0.f;
#pragma unroll
    for (int d = 0; d < 8; ++d) { qq[d] = (float)row[h * 8 + d]; s += qq[d] * qq[d]; }
    float inv = 1.0f / fmaxf(sqrtf(s), 1e-12f);
#pragma unroll
    for (int d = 0; d < 8; ++d) qq[d] *= inv;

    float den = (float)kT;
#pragma unroll
    for (int i = 0; i < 8; ++i) den += qq[i] * s_ksum[h][i];
    float invden = 1.0f / fmaxf(den, 1e-5f);

    _Float16* orow = outcat + (((size_t)(b * kT + t)) * kN + n) * 128 + 64 + h * 8;
#pragma unroll
    for (int d = 0; d < 8; ++d) {
      float num = (float)kT * s_v[t][h][d];
#pragma unroll
      for (int i = 0; i < 8; ++i) num += qq[i] * s_kvs[h][i][d];
      orow[d] = (_Float16)(num * invden);
    }
  }
}

// =====================================================================
// fused x_glo + LayerNorm1 : y = LN(2*(x + att0*p0 + att1*p1*0.01))
// one block = 4 rows of 64
// =====================================================================
__global__ void glo_ln1(const float* __restrict__ x, const float* __restrict__ att0,
                        const float* __restrict__ att1, const float* __restrict__ p0,
                        const float* __restrict__ p1, const float* __restrict__ g,
                        const float* __restrict__ bta,
                        float* __restrict__ y32, _Float16* __restrict__ y16) {
  const int row = threadIdx.x >> 6;
  const int c   = threadIdx.x & 63;
  const size_t r = (size_t)blockIdx.x * 4 + row;
  const size_t idx = r * 64 + c;
  float gv = 2.0f * (x[idx] + att0[idx] * p0[idx] + att1[idx] * p1[idx] * 0.01f);
  __shared__ float s1[4][64];
  __shared__ float s2[4][64];
  s1[row][c] = gv;
  s2[row][c] = gv * gv;
  __syncthreads();
  for (int o = 32; o > 0; o >>= 1) {
    if (c < o) { s1[row][c] += s1[row][c + o]; s2[row][c] += s2[row][c + o]; }
    __syncthreads();
  }
  const float m   = s1[row][0] * (1.0f / 64.0f);
  const float var = s2[row][0] * (1.0f / 64.0f) - m * m;
  const float out = (gv - m) * rsqrtf(var + 1e-5f) * g[c] + bta[c];
  y32[idx] = out;
  y16[idx] = (_Float16)out;
}

// final: out = LN(y + h)
__global__ void add_ln2(const float* __restrict__ y, const float* __restrict__ hmlp,
                        const float* __restrict__ g, const float* __restrict__ bta,
                        float* __restrict__ out) {
  const int row = threadIdx.x >> 6;
  const int c   = threadIdx.x & 63;
  const size_t r = (size_t)blockIdx.x * 4 + row;
  const size_t idx = r * 64 + c;
  float gv = y[idx] + hmlp[idx];
  __shared__ float s1[4][64];
  __shared__ float s2[4][64];
  s1[row][c] = gv;
  s2[row][c] = gv * gv;
  __syncthreads();
  for (int o = 32; o > 0; o >>= 1) {
    if (c < o) { s1[row][c] += s1[row][c + o]; s2[row][c] += s2[row][c + o]; }
    __syncthreads();
  }
  const float m   = s1[row][0] * (1.0f / 64.0f);
  const float var = s2[row][0] * (1.0f / 64.0f) - m * m;
  out[idx] = (gv - m) * rsqrtf(var + 1e-5f) * g[c] + bta[c];
}

// =====================================================================
// launcher
// =====================================================================
extern "C" void kernel_launch(void* const* d_in, const int* in_sizes, int n_in,
                              void* d_out, int out_size, void* d_ws, size_t ws_size,
                              hipStream_t stream) {
  const float* x       = (const float*)d_in[0];
  const float* phi     = (const float*)d_in[1];
  const float* phi_inv = (const float*)d_in[2];
  const float* diag_w  = (const float*)d_in[3];
  const float* qkv0_w  = (const float*)d_in[4];
  const float* out0_w  = (const float*)d_in[5];
  const float* out0_b  = (const float*)d_in[6];
  const float* qkv1_w  = (const float*)d_in[7];
  const float* out1_w  = (const float*)d_in[8];
  const float* out1_b  = (const float*)d_in[9];
  const float* pw0_w   = (const float*)d_in[10];
  const float* pw0_b   = (const float*)d_in[11];
  const float* pw1_w   = (const float*)d_in[12];
  const float* pw1_b   = (const float*)d_in[13];
  const float* fc1_w   = (const float*)d_in[14];
  const float* fc1_b   = (const float*)d_in[15];
  const float* fc2_w   = (const float*)d_in[16];
  const float* fc2_b   = (const float*)d_in[17];
  const float* ln1_g   = (const float*)d_in[18];
  const float* ln1_b   = (const float*)d_in[19];
  const float* ln2_g   = (const float*)d_in[20];
  const float* ln2_b   = (const float*)d_in[21];
  float* outp = (float*)d_out;

  char* ws = (char*)d_ws;
  size_t cur = 0;
  auto alloc = [&](size_t bytes) {
    size_t o = cur;
    cur += (bytes + 255) & ~(size_t)255;
    return o;
  };

  // weights (f16)
  const size_t o_wq0 = alloc(192 * 64 * 2);
  const size_t o_wo0 = alloc(64 * 128 * 2);
  const size_t o_wq1 = alloc(192 * 64 * 2);
  const size_t o_wo1 = alloc(64 * 128 * 2);
  const size_t o_wp0 = alloc(64 * 64 * 2);
  const size_t o_wp1 = alloc(64 * 64 * 2);
  const size_t o_wf1 = alloc(128 * 64 * 2);
  const size_t o_wf2 = alloc(64 * 128 * 2);
  // persistent buffers
  const size_t o_msum = alloc((size_t)kN * kN * 2);            // Msum f16
  const size_t o_xh   = alloc((size_t)kBTNC * 2);              // x row-major f16
  const size_t o_exk  = alloc((size_t)kBTNC * 2);              // xk f16 -> later y f16
  const size_t o_a0f  = alloc((size_t)kBTNC * 4);              // att0 f32
  const size_t o_a0h  = alloc((size_t)kBTNC * 2);              // att0 f16
  const size_t o_a1f  = alloc((size_t)kBTNC * 4);              // att1 f32
  const size_t o_y32  = alloc((size_t)kBTNC * 4);              // y f32
  // phase-reused regions
  const size_t o_regA = alloc((size_t)kBTN * 192 * 2);         // qkv f16 -> p0 f32 -> h1 f16
  const size_t o_regB = alloc((size_t)kBTN * 128 * 2);         // Ah+Bth -> xt -> outcat -> p1 -> h2
  if (cur > ws_size) return;  // not enough scratch: bail deterministically

  _Float16* wq0 = (_Float16*)(ws + o_wq0);
  _Float16* wo0 = (_Float16*)(ws + o_wo0);
  _Float16* wq1 = (_Float16*)(ws + o_wq1);
  _Float16* wo1 = (_Float16*)(ws + o_wo1);
  _Float16* wp0 = (_Float16*)(ws + o_wp0);
  _Float16* wp1 = (_Float16*)(ws + o_wp1);
  _Float16* wf1 = (_Float16*)(ws + o_wf1);
  _Float16* wf2 = (_Float16*)(ws + o_wf2);
  _Float16* msumh  = (_Float16*)(ws + o_msum);
  _Float16* xhrow  = (_Float16*)(ws + o_xh);
  _Float16* xkh    = (_Float16*)(ws + o_exk);
  _Float16* yh     = (_Float16*)(ws + o_exk);
  float*    att0f  = (float*)(ws + o_a0f);
  _Float16* att0h  = (_Float16*)(ws + o_a0h);
  float*    att1f  = (float*)(ws + o_a1f);
  float*    y32    = (float*)(ws + o_y32);
  _Float16* qkvh   = (_Float16*)(ws + o_regA);
  float*    p0     = (float*)(ws + o_regA);
  _Float16* h1h    = (_Float16*)(ws + o_regA);
  _Float16* Ah     = (_Float16*)(ws + o_regB);
  _Float16* Bth    = (_Float16*)(ws + o_regB + (size_t)kN * kKBIG * 2);
  _Float16* xth    = (_Float16*)(ws + o_regB);
  _Float16* outcat = (_Float16*)(ws + o_regB);
  float*    p1     = (float*)(ws + o_regB);
  float*    h2     = (float*)(ws + o_regB);

  // ---- 1. weight converts ----
  cvt_f32_f16<<<48, 256, 0, stream>>>(qkv0_w, wq0, 192 * 64);
  cvt_f32_f16<<<32, 256, 0, stream>>>(out0_w, wo0, 64 * 128);
  cvt_f32_f16<<<48, 256, 0, stream>>>(qkv1_w, wq1, 192 * 64);
  cvt_f32_f16<<<32, 256, 0, stream>>>(out1_w, wo1, 64 * 128);
  cvt_f32_f16<<<16, 256, 0, stream>>>(pw0_w, wp0, 64 * 64);
  cvt_f32_f16<<<16, 256, 0, stream>>>(pw1_w, wp1, 64 * 64);
  cvt_f32_f16<<<32, 256, 0, stream>>>(fc1_w, wf1, 128 * 64);
  cvt_f32_f16<<<32, 256, 0, stream>>>(fc2_w, wf2, 64 * 128);

  // ---- 2. build Msum operands, run big GEMM ----
  build_ah<<<(kS * kN * kN) / 256, 256, 0, stream>>>(phi, Ah);
  build_bth<<<dim3(kN / 32, kN / 32, kS), dim3(32, 8), 0, stream>>>(phi_inv, diag_w, Bth);
  gemm_wmma<2, 4, 4, 2, false, false, false, true>
      <<<dim3(kN / 128, kN / 128, 1), 256, 0, stream>>>(
          Ah, kKBIG, 0, Bth, kKBIG, 0, nullptr, msumh, kN, 0, nullptr, kKBIG);

  // ---- 3. x layouts, batched xk GEMM ----
  cvt_f32_f16<<<(unsigned)(kBTNC / 256), 256, 0, stream>>>(x, xhrow, kBTNC);
  build_xt<<<dim3(kN / 32, kC / 32, kBT), dim3(32, 8), 0, stream>>>(x, xth);
  gemm_wmma<2, 2, 4, 2, false, false, false, true>
      <<<dim3(kN / 128, 1, kBT), 256, 0, stream>>>(
          msumh, kN, 0, xth, kN, (long long)kC * kN,
          nullptr, xkh, kC, (long long)kN * kC, nullptr, kN);

  // ---- 4. layer 0: qkv, attention, projection ----
  gemm_wmma<2, 2, 4, 2, false, false, false, true>
      <<<dim3(kBTN / 128, 3, 1), 256, 0, stream>>>(
          xhrow, 64, 0, wq0, 64, 0, nullptr, qkvh, 192, 0, nullptr, 64);
  attn_spatial<<<dim3(kBT, kH), 256, 0, stream>>>(qkvh, outcat);
  attn_temporal<<<kB * kN, 192, 0, stream>>>(qkvh, outcat);
  gemm_wmma<2, 2, 4, 2, true, false, true, true>
      <<<dim3(kBTN / 128, 1, 1), 256, 0, stream>>>(
          outcat, 128, 0, wo0, 128, 0, att0f, att0h, 64, 0, out0_b, 128);

  // ---- 5. layer 1: qkv (from xk), attention, projection ----
  gemm_wmma<2, 2, 4, 2, false, false, false, true>
      <<<dim3(kBTN / 128, 3, 1), 256, 0, stream>>>(
          xkh, 64, 0, wq1, 64, 0, nullptr, qkvh, 192, 0, nullptr, 64);
  attn_spatial<<<dim3(kBT, kH), 256, 0, stream>>>(qkvh, outcat);
  attn_temporal<<<kB * kN, 192, 0, stream>>>(qkvh, outcat);
  gemm_wmma<2, 2, 4, 2, true, false, true, false>
      <<<dim3(kBTN / 128, 1, 1), 256, 0, stream>>>(
          outcat, 128, 0, wo1, 128, 0, att1f, nullptr, 64, 0, out1_b, 128);

  // ---- 6. pw projections ----
  gemm_wmma<2, 2, 4, 2, true, false, true, false>
      <<<dim3(kBTN / 128, 1, 1), 256, 0, stream>>>(
          xhrow, 64, 0, wp0, 64, 0, p0, nullptr, 64, 0, pw0_b, 64);
  gemm_wmma<2, 2, 4, 2, true, false, true, false>
      <<<dim3(kBTN / 128, 1, 1), 256, 0, stream>>>(
          att0h, 64, 0, wp1, 64, 0, p1, nullptr, 64, 0, pw1_b, 64);

  // ---- 7. x_glo + LN1 ----
  glo_ln1<<<kBTN / 4, 256, 0, stream>>>(x, att0f, att1f, p0, p1, ln1_g, ln1_b, y32, yh);

  // ---- 8. MLP ----
  gemm_wmma<2, 2, 4, 2, true, true, false, true>
      <<<dim3(kBTN / 128, 2, 1), 256, 0, stream>>>(
          yh, 64, 0, wf1, 64, 0, nullptr, h1h, 128, 0, fc1_b, 64);
  gemm_wmma<2, 2, 4, 2, true, false, true, false>
      <<<dim3(kBTN / 128, 1, 1), 256, 0, stream>>>(
          h1h, 128, 0, wf2, 128, 0, h2, nullptr, 64, 0, fc2_b, 128);

  // ---- 9. LN2 -> output ----
  add_ln2<<<kBTN / 4, 256, 0, stream>>>(y32, h2, ln2_g, ln2_b, outp);
}
